// CNN_LSTM_CRF_6382321402326
// MI455X (gfx1250) — compile-verified, matrix operand order checked by reading, not run on hardware
//
#include <hip/hip_runtime.h>
#include <cstdint>
#include <cstddef>

// ---------------------------------------------------------------------------
// CNN-LSTM-CRF for MI455X (gfx1250, wave32, WMMA).
// All GEMMs use v_wmma_f32_16x16x32_bf16 with the documented CDNA5 wave32
// fragment swizzles. Recurrent step fuses [x_t | h_{t-1}] @ [W_ih | W_hh]^T
// (K = 320 + 512 = 832, zero-padded weight columns absorb the x-row overreads),
// and each wave computes all four gate tiles (i,f,g,o) for one 16-wide hidden
// column so the LSTM cell update happens in accumulator registers (no LDS).
// ---------------------------------------------------------------------------

typedef __bf16 bf16;
typedef __attribute__((ext_vector_type(8)))  __bf16 v8bf;
typedef __attribute__((ext_vector_type(16))) __bf16 v16bf;
typedef __attribute__((ext_vector_type(8)))  float  v8f;

#define BB 256   // batch
#define LL 180   // seq len
#define EE 300   // embed dim
#define EP 320   // embed dim padded (16B-aligned rows, mult of 32)
#define HH 512   // hidden
#define TT 31    // tags
#define LP 192   // L padded to mult of 32 (conv K dim)
#define KC 832   // fused LSTM K: 320 (x region) + 512 (h region)
#define NG 2048  // 4*H gates

union Frag { v16bf v; v8bf h[2]; };

__device__ __forceinline__ float sigf(float x) { return 1.f / (1.f + __expf(-x)); }
__device__ __forceinline__ float tanhfast(float x) { return 1.f - 2.f / (__expf(2.f * x) + 1.f); }

// Load a 16x32 bf16 fragment (A layout; also B when memory is N-major):
// lane&15 selects the row, lane>>4 selects the {0,8} half-K offset; each half
// is 8 contiguous bf16 = one 16-byte global_load_b128.
__device__ __forceinline__ v16bf load_frag(const bf16* rowbase, int kb) {
  Frag f;
  f.h[0] = *(const v8bf*)(rowbase + kb);
  f.h[1] = *(const v8bf*)(rowbase + kb + 16);
  return f.v;
}

// --------------------------- small prep kernels ----------------------------

__global__ void k_zero_f32(float* p, int n) {
  int i = blockIdx.x * 256 + threadIdx.x;
  if (i < n) p[i] = 0.f;
}
__global__ void k_zero_bf16(bf16* p, int n) {
  int i = blockIdx.x * 256 + threadIdx.x;
  if (i < n) p[i] = (bf16)0.f;
}

// Wcat[n][k] (bf16, [2048 x 832] N-major): k<300 -> w_ih[n,k]; 320<=k -> w_hh[n,k-320];
// zeros elsewhere (they neutralize the aligned x-row overreads).
__global__ void k_pack_wcat(const float* __restrict__ wih, const float* __restrict__ whh,
                            bf16* __restrict__ wcat) {
  int i = blockIdx.x * 256 + threadIdx.x;
  if (i >= NG * KC) return;
  int n = i / KC, k = i % KC;
  float v = 0.f;
  if (k < EE) v = wih[(size_t)n * EE + k];
  else if (k >= EP) v = whh[(size_t)n * HH + (k - EP)];
  wcat[i] = (bf16)v;
}

__global__ void k_pack_cls(const float* __restrict__ clsw, bf16* __restrict__ clsp) {
  int i = blockIdx.x * 256 + threadIdx.x;
  if (i >= 32 * 1024) return;
  int n = i >> 10, k = i & 1023;
  clsp[i] = (bf16)((n < TT) ? clsw[(size_t)n * 1024 + k] : 0.f);
}

// Ac[tap][l_out][l_in] bf16, padded 192x192 (zeros beyond 180).
__global__ void k_pack_ac(const float* __restrict__ convw, bf16* __restrict__ ac) {
  int i = blockIdx.x * 256 + threadIdx.x;
  if (i >= 3 * LP * LP) return;
  int tap = i / (LP * LP);
  int r = (i / LP) % LP, c = i % LP;
  ac[i] = (bf16)((r < LL && c < LL) ? convw[((size_t)r * LL + c) * 3 + tap] : 0.f);
}

// embT[b][e][l] bf16 (L padded to 192 with zeros): transposed so the conv's
// B operand (K = l_in contiguous) loads with the standard fragment pattern.
__global__ void k_gather_embT(const int* __restrict__ inputs, const float* __restrict__ table,
                              bf16* __restrict__ embT) {
  int i = blockIdx.x * 256 + threadIdx.x;
  if (i >= BB * EE * LP) return;
  int l = i % LP;
  int e = (i / LP) % EE;
  int b = i / (LP * EE);
  float v = 0.f;
  if (l < LL) {
    int tok = inputs[b * LL + l];
    v = table[(size_t)tok * EE + e];
  }
  embT[i] = (bf16)v;
}

// ------------------------------- conv GEMM ---------------------------------
// x[b, l_out, e] = sum_{l_in,tap} Ac[tap][l_out][l_in] * emb[b, l_in, e+tap-1] + conv_b[l_out]
// Block: 128 thr = 4 waves; block tile M=16 (l_out), N=64 (e), wave owns 16 cols.
__global__ void k_conv(const bf16* __restrict__ embT, const bf16* __restrict__ ac,
                       const float* __restrict__ convb, bf16* __restrict__ x) {
  int b = blockIdx.z;
  int m0 = blockIdx.y * 16;
  int wave = threadIdx.x >> 5, lane = threadIdx.x & 31;
  int e0 = blockIdx.x * 64 + wave * 16;
  int r = lane & 15;
  int kshift = (lane >> 4) << 3;  // 0 or 8
  v8f acc = {};
  for (int tap = 0; tap < 3; ++tap) {
    const bf16* A = ac + (size_t)tap * LP * LP + (size_t)(m0 + r) * LP;
    int eidx = e0 + r + tap - 1;                  // shifted e row for this lane's N column
    bool inb = (eidx >= 0 && eidx < EE);
    const bf16* brow = embT + ((size_t)b * EE + (inb ? eidx : 0)) * LP;
    for (int kt = 0; kt < 6; ++kt) {
      int kb = kt * 32 + kshift;
      Frag fa, fb;
      fa.v = load_frag(A, kb);
      if (inb) {
        fb.v = load_frag(brow, kb);
      } else {
        v8bf z = {};
        fb.h[0] = z; fb.h[1] = z;
      }
      acc = __builtin_amdgcn_wmma_f32_16x16x32_bf16(false, fa.v, false, fb.v,
                                                    (short)0, acc, false, false);
    }
  }
  int e = e0 + (lane & 15);
#pragma unroll
  for (int rr = 0; rr < 8; ++rr) {
    int m = m0 + rr + kshift;  // C layout: lanes>=16 hold rows +8
    if (m < LL && e < EP) {
      float v = (e < EE) ? (acc[rr] + convb[m]) : 0.f;  // zero the pad cols
      x[((size_t)b * LL + m) * EP + e] = (bf16)v;
    }
  }
}

// ------------------------------ LSTM step ----------------------------------
// gates[b, n] = sum_k [x_t | h_prev](b,k) * Wcat(n,k);  then cell update.
// Block: 128 thr = 4 waves. Wave owns a 16(batch) x 16(hidden) cell tile and
// accumulates the i,f,g,o gate tiles for it (4 WMMAs share one A fragment),
// so the cell update runs straight out of the accumulators: no LDS, no
// barrier. Ping-pong h buffers make the cross-block h RAW hazard a
// kernel-boundary dependency.
__global__ void k_lstm_step(const bf16* __restrict__ x, const bf16* __restrict__ wcat,
                            const float* __restrict__ bias,
                            const bf16* __restrict__ hprev, bf16* __restrict__ hnext,
                            float* __restrict__ cbuf, bf16* __restrict__ hseq, int t) {
  int m0 = blockIdx.x * 16;                 // batch tile
  int wave = threadIdx.x >> 5, lane = threadIdx.x & 31;
  int u0 = blockIdx.y * 64 + wave * 16;     // hidden tile owned by this wave
  int r = lane & 15;
  int kshift = (lane >> 4) << 3;
  const bf16* xrow = x + ((size_t)(m0 + r) * LL + t) * EP;   // 16B-aligned (EP*2 bytes)
  const bf16* hrow = hprev + (size_t)(m0 + r) * HH;
  const bf16* w0 = wcat + (size_t)(0 * HH + u0 + r) * KC;    // i
  const bf16* w1 = wcat + (size_t)(1 * HH + u0 + r) * KC;    // f
  const bf16* w2 = wcat + (size_t)(2 * HH + u0 + r) * KC;    // g
  const bf16* w3 = wcat + (size_t)(3 * HH + u0 + r) * KC;    // o
  v8f ai = {}, af = {}, ag = {}, ao = {};
#pragma unroll 2
  for (int kt = 0; kt < 26; ++kt) {
    int kb = kt * 32 + kshift;
    Frag fa, b0, b1, b2, b3;
#pragma unroll
    for (int hh = 0; hh < 2; ++hh) {
      int ks = kb + hh * 16;
      const bf16* p = (ks < EP) ? (xrow + ks) : (hrow + (ks - EP));
      fa.h[hh] = *(const v8bf*)p;
    }
    b0.v = load_frag(w0, kb);
    b1.v = load_frag(w1, kb);
    b2.v = load_frag(w2, kb);
    b3.v = load_frag(w3, kb);
    __builtin_prefetch(w0 + kb + 256, 0, 1);  // stream weights (global_prefetch_b8)
    ai = __builtin_amdgcn_wmma_f32_16x16x32_bf16(false, fa.v, false, b0.v, (short)0, ai, false, false);
    af = __builtin_amdgcn_wmma_f32_16x16x32_bf16(false, fa.v, false, b1.v, (short)0, af, false, false);
    ag = __builtin_amdgcn_wmma_f32_16x16x32_bf16(false, fa.v, false, b2.v, (short)0, ag, false, false);
    ao = __builtin_amdgcn_wmma_f32_16x16x32_bf16(false, fa.v, false, b3.v, (short)0, ao, false, false);
  }
  // Cell update straight from accumulators: lane owns column u for 8 rows.
  int u = u0 + (lane & 15);
  float bi = bias[u];
  float bf_ = bias[HH + u];
  float bg = bias[2 * HH + u];
  float bo = bias[3 * HH + u];
#pragma unroll
  for (int rr = 0; rr < 8; ++rr) {
    int brow = m0 + rr + kshift;  // C layout: lanes>=16 hold rows +8
    size_t ci = (size_t)brow * HH + u;
    float gi = ai[rr] + bi;
    float gf = af[rr] + bf_;
    float gg = ag[rr] + bg;
    float go = ao[rr] + bo;
    float c = sigf(gf) * cbuf[ci] + sigf(gi) * tanhfast(gg);
    cbuf[ci] = c;
    float h = sigf(go) * tanhfast(c);
    hnext[ci] = (bf16)h;
    hseq[((size_t)brow * LL + t) * HH + u] = (bf16)h;
  }
}

// ------------------------------ classifier ---------------------------------
// em[r, tt] = [hf(r) | hb(r)] @ cls_w[tt]^T + cls_b[tt];  M = B*L, N = 32 (pad), K = 1024.
__global__ void k_cls(const bf16* __restrict__ hf, const bf16* __restrict__ hb,
                      const bf16* __restrict__ clsp, const float* __restrict__ clsb,
                      float* __restrict__ em) {
  int wave = threadIdx.x >> 5, lane = threadIdx.x & 31;
  int row0 = blockIdx.x * 64 + wave * 16;
  int r = lane & 15;
  int kshift = (lane >> 4) << 3;
  const bf16* fr = hf + (size_t)(row0 + r) * HH;
  const bf16* br = hb + (size_t)(row0 + r) * HH;
  const bf16* q0 = clsp + (size_t)r * 1024;
  const bf16* q1 = clsp + (size_t)(16 + r) * 1024;
  v8f acc0 = {}, acc1 = {};
  for (int kt = 0; kt < 32; ++kt) {
    int kb = kt * 32 + kshift;
    Frag fa, fb0, fb1;
#pragma unroll
    for (int hh = 0; hh < 2; ++hh) {
      int ks = kb + hh * 16;
      const bf16* p = (ks < HH) ? (fr + ks) : (br + (ks - HH));
      fa.h[hh] = *(const v8bf*)p;
    }
    fb0.v = load_frag(q0, kb);
    fb1.v = load_frag(q1, kb);
    acc0 = __builtin_amdgcn_wmma_f32_16x16x32_bf16(false, fa.v, false, fb0.v,
                                                   (short)0, acc0, false, false);
    acc1 = __builtin_amdgcn_wmma_f32_16x16x32_bf16(false, fa.v, false, fb1.v,
                                                   (short)0, acc1, false, false);
  }
#pragma unroll
  for (int rr = 0; rr < 8; ++rr) {
    int m = row0 + rr + kshift;
    int n = lane & 15;
    em[(size_t)m * 32 + n] = acc0[rr] + clsb[n];  // n < 16 < 31 always valid
    int n1 = 16 + n;
    em[(size_t)m * 32 + n1] = (n1 < TT) ? (acc1[rr] + clsb[n1]) : -1e30f;
  }
}

// ------------------------------- Viterbi -----------------------------------
// One wave32 per batch; lane j owns tag j (T=31). trans staged in LDS,
// score broadcast via wave shuffles. Lane 0 backtracks.
__global__ void k_viterbi(const float* __restrict__ em, const int* __restrict__ mask,
                          const float* __restrict__ startt, const float* __restrict__ endt,
                          const float* __restrict__ trans, unsigned char* __restrict__ hist,
                          float* __restrict__ outp) {
  int b = blockIdx.x;
  int j = threadIdx.x;  // 0..31
  __shared__ float tl[31 * 32];
  for (int p = j; p < 31 * 32; p += 32) {
    int i = p >> 5, jj = p & 31;
    tl[p] = (jj < TT) ? trans[i * TT + jj] : 0.f;
  }
  __syncthreads();
  const float NEG = -1e30f;
  float score = (j < TT) ? (startt[j] + em[(size_t)b * LL * 32 + j]) : NEG;
  for (int t = 1; t < LL; ++t) {
    float emj = em[((size_t)b * LL + t) * 32 + j];
    float best = NEG;
    int arg = 0;
    for (int i = 0; i < TT; ++i) {
      float si = __shfl(score, i, 32);
      float cand = si + tl[i * 32 + j];
      if (cand > best) { best = cand; arg = i; }  // first-max like argmax
    }
    best += emj;
    if (mask[b * LL + t]) score = (j < TT) ? best : NEG;
    hist[((size_t)(t - 1) * BB + b) * 32 + j] = (unsigned char)arg;
  }
  float fin = (j < TT) ? (score + endt[j]) : NEG;
  float bestf = NEG;
  int last = 0;
  for (int i = 0; i < TT; ++i) {
    float v = __shfl(fin, i, 32);
    if (v > bestf) { bestf = v; last = i; }
  }
  if (j == 0) {
    outp[BB * LL + b] = bestf;
    int tag = last;
    outp[b * LL + (LL - 1)] = (float)tag;
    for (int t = LL - 2; t >= 0; --t) {
      int prev = hist[((size_t)t * BB + b) * 32 + tag];
      if (mask[b * LL + t + 1]) tag = prev;
      outp[b * LL + t] = (float)tag;
    }
  }
}

// ------------------------------- launcher ----------------------------------

extern "C" void kernel_launch(void* const* d_in, const int* in_sizes, int n_in,
                              void* d_out, int out_size, void* d_ws, size_t ws_size,
                              hipStream_t stream) {
  (void)in_sizes; (void)n_in; (void)out_size; (void)ws_size;
  const int*   inputs = (const int*)d_in[0];
  const int*   mask   = (const int*)d_in[1];
  const float* table  = (const float*)d_in[2];
  const float* convw  = (const float*)d_in[3];
  const float* convb  = (const float*)d_in[4];
  const float* wihf   = (const float*)d_in[5];
  const float* whhf   = (const float*)d_in[6];
  const float* biasf  = (const float*)d_in[7];
  const float* wihb   = (const float*)d_in[8];
  const float* whhb   = (const float*)d_in[9];
  const float* biasb  = (const float*)d_in[10];
  const float* clsw   = (const float*)d_in[11];
  const float* clsb   = (const float*)d_in[12];
  const float* stt    = (const float*)d_in[13];
  const float* ent    = (const float*)d_in[14];
  const float* trn    = (const float*)d_in[15];
  float* out = (float*)d_out;

  char* ws = (char*)d_ws;
  size_t off = 0;
  auto take = [&](size_t bytes) -> char* {
    char* p = ws + off;
    off = (off + bytes + 255) & ~(size_t)255;
    return p;
  };
  bf16*  x     = (bf16*)take((size_t)BB * LL * EP * 2);   // conv output, padded rows
  bf16*  wcf   = (bf16*)take((size_t)NG * KC * 2);
  bf16*  wcb   = (bf16*)take((size_t)NG * KC * 2);
  bf16*  clsp  = (bf16*)take((size_t)32 * 1024 * 2);
  bf16*  ac    = (bf16*)take((size_t)3 * LP * LP * 2);
  bf16*  hAf   = (bf16*)take((size_t)BB * HH * 2);
  bf16*  hBf   = (bf16*)take((size_t)BB * HH * 2);
  bf16*  hAb   = (bf16*)take((size_t)BB * HH * 2);
  bf16*  hBb   = (bf16*)take((size_t)BB * HH * 2);
  float* cbf   = (float*)take((size_t)BB * HH * 4);
  float* cbb   = (float*)take((size_t)BB * HH * 4);
  bf16*  hseqf = (bf16*)take((size_t)BB * LL * HH * 2);
  bf16*  hseqb = (bf16*)take((size_t)BB * LL * HH * 2);
  float* em    = (float*)take((size_t)BB * LL * 32 * 4);
  unsigned char* hist = (unsigned char*)take((size_t)(LL - 1) * BB * 32);
  bf16*  embT  = (bf16*)hseqf;  // alias: embT dead once conv completes, hseqf written later

  // --- prep / repack (stream-ordered) ---
  k_pack_ac<<<(3 * LP * LP + 255) / 256, 256, 0, stream>>>(convw, ac);
  k_pack_wcat<<<(NG * KC + 255) / 256, 256, 0, stream>>>(wihf, whhf, wcf);
  k_pack_wcat<<<(NG * KC + 255) / 256, 256, 0, stream>>>(wihb, whhb, wcb);
  k_pack_cls<<<(32 * 1024 + 255) / 256, 256, 0, stream>>>(clsw, clsp);
  k_gather_embT<<<(BB * EE * LP + 255) / 256, 256, 0, stream>>>(inputs, table, embT);
  k_zero_bf16<<<(BB * HH + 255) / 256, 256, 0, stream>>>(hAf, BB * HH);
  k_zero_bf16<<<(BB * HH + 255) / 256, 256, 0, stream>>>(hAb, BB * HH);
  k_zero_f32<<<(BB * HH + 255) / 256, 256, 0, stream>>>(cbf, BB * HH);
  k_zero_f32<<<(BB * HH + 255) / 256, 256, 0, stream>>>(cbb, BB * HH);

  // --- conv over embedding dim (WMMA) ---
  k_conv<<<dim3(EP / 64, LP / 16, BB), 128, 0, stream>>>(embT, ac, convb, x);

  // --- BiLSTM: 180 dependent steps, fwd and bwd interleaved on the stream ---
  bf16* hpf = hAf; bf16* hnf = hBf;
  bf16* hpb = hAb; bf16* hnb = hBb;
  for (int s = 0; s < LL; ++s) {
    k_lstm_step<<<dim3(BB / 16, HH / 64), 128, 0, stream>>>(x, wcf, biasf, hpf, hnf, cbf, hseqf, s);
    k_lstm_step<<<dim3(BB / 16, HH / 64), 128, 0, stream>>>(x, wcb, biasb, hpb, hnb, cbb, hseqb, LL - 1 - s);
    bf16* t0 = hpf; hpf = hnf; hnf = t0;
    bf16* t1 = hpb; hpb = hnb; hnb = t1;
  }

  // --- classifier (emissions) ---
  k_cls<<<(BB * LL) / 64, 128, 0, stream>>>(hseqf, hseqb, clsp, clsb, em);

  // --- CRF Viterbi decode + backtrack ---
  k_viterbi<<<BB, 32, 0, stream>>>(em, mask, stt, ent, trn, hist, out);
}
